// stackedIndRNN_encoder_58677843198245
// MI455X (gfx1250) — matrix operand order repaired and checked by reading, not compiled, not run on hardware
//
#include <hip/hip_runtime.h>

// ---------------- problem constants ----------------
#define T_DIM 2048
#define B_DIM 32
#define D_DIM 128
#define H_DIM 512
#define L_DIM 6
#define C_DIM 10
#define M_DIM (T_DIM * B_DIM)        // 65536 rows for the big GEMMs
#define NB    (B_DIM * H_DIM)        // 16384 elements per timestep
#define BN_EPS 1e-5f
#define INV_N (1.0f / (float)(T_DIM * B_DIM))

typedef __bf16 bf16;
typedef __attribute__((ext_vector_type(16))) __bf16 v16bf;
typedef __attribute__((ext_vector_type(8)))  __bf16 v8bf;
typedef __attribute__((ext_vector_type(4)))  __bf16 v4bf;
typedef __attribute__((ext_vector_type(8)))  float  v8f;

// ---------------- fp32 -> bf16 hi/lo split of x ----------------
__global__ __launch_bounds__(256) void convx_kernel(const float* __restrict__ x,
                                                    bf16* __restrict__ xHi,
                                                    bf16* __restrict__ xLo) {
  size_t i = ((size_t)blockIdx.x * 256 + threadIdx.x) * 4;
  float4 v = *(const float4*)(x + i);
  bf16 h0 = (bf16)v.x, h1 = (bf16)v.y, h2 = (bf16)v.z, h3 = (bf16)v.w;
  v4bf hv = {h0, h1, h2, h3};
  v4bf lv = {(bf16)(v.x - (float)h0), (bf16)(v.y - (float)h1),
             (bf16)(v.z - (float)h2), (bf16)(v.w - (float)h3)};
  *(v4bf*)(xHi + i) = hv;
  *(v4bf*)(xLo + i) = lv;
}

// ---------------- per-layer weight prep: fold previous BN into W, split to bf16 hi/lo ----------------
// a[i] = gamma_prev[i] * rsqrt(var+eps); c[i] = beta_prev[i] - mean*a[i]
// Wadj[n,i] = W[n,i]*a[i];  badj[n] = b[n] + sum_i W[n,i]*c[i]
__global__ __launch_bounds__(256) void prep_kernel(
    const float* __restrict__ W, const float* __restrict__ bvec,
    const float* __restrict__ gammaPrev, const float* __restrict__ betaPrev,
    const float* __restrict__ partSum, const float* __restrict__ partSq,
    bf16* __restrict__ wHi, bf16* __restrict__ wLo, float* __restrict__ badj,
    int Din, int useBN) {
  __shared__ float sA[H_DIM];
  __shared__ float sC[H_DIM];
  __shared__ float red[256];
  const int n = blockIdx.x;
  const int t = threadIdx.x;
  for (int i = t; i < Din; i += 256) {
    float a = 1.f, c = 0.f;
    if (useBN) {
      float s = 0.f, q = 0.f;
      for (int bb = 0; bb < B_DIM; ++bb) {          // fixed-order deterministic reduce
        s += partSum[bb * H_DIM + i];
        q += partSq[bb * H_DIM + i];
      }
      float mean = s * INV_N;
      float var  = q * INV_N - mean * mean;
      a = gammaPrev[i] * rsqrtf(var + BN_EPS);
      c = betaPrev[i] - mean * a;
    }
    sA[i] = a; sC[i] = c;
  }
  __syncthreads();
  const float* wrow = W + (size_t)n * Din;
  bf16* hrow = wHi + (size_t)n * Din;
  bf16* lrow = wLo + (size_t)n * Din;
  float dot = 0.f;
  for (int i = t; i < Din; i += 256) {
    float w  = wrow[i];
    float wa = w * sA[i];
    bf16 hi = (bf16)wa;
    hrow[i] = hi;
    lrow[i] = (bf16)(wa - (float)hi);
    dot += w * sC[i];
  }
  red[t] = dot;
  __syncthreads();
  for (int s = 128; s > 0; s >>= 1) {
    if (t < s) red[t] += red[t + s];
    __syncthreads();
  }
  if (t == 0) badj[n] = bvec[n] + red[0];
}

// ---------------- split-bf16 WMMA GEMM: z[M,512] = A[M,K] * Wadj[512,K]^T + badj ----------------
// K is a compile-time constant: full k-unroll -> all loads become base-ptr + immediate
// offset (no in-loop address VALU -> no WMMA WAR hazard NOPs).
template <int K>
__global__ __launch_bounds__(256) void gemm_kernel(
    const bf16* __restrict__ aHip, const bf16* __restrict__ aLop,
    const bf16* __restrict__ wHip, const bf16* __restrict__ wLop,
    const float* __restrict__ badj, float* __restrict__ z) {
  const int lane  = threadIdx.x & 31;
  const int wave  = threadIdx.x >> 5;
  const int waveM = wave & 1;              // 2 waves along M
  const int waveN = wave >> 1;             // 4 waves along N
  const int mOff  = blockIdx.x * 64 + waveM * 32;
  const int nOff  = blockIdx.y * 256 + waveN * 64;
  const int half  = lane >> 4;             // lane group 0/1
  const int lr    = lane & 15;
  const int aC0   = half * 16;             // A: 16 contiguous k per lane group (matches B)
  const int bC0   = half * 16;             // B: 16 contiguous k per lane group

  // loop-invariant per-row base pointers (unrolled loads use immediate offsets)
  const bf16* aHr[2]; const bf16* aLr[2];
  const bf16* wHr[4]; const bf16* wLr[4];
#pragma unroll
  for (int mi = 0; mi < 2; ++mi) {
    size_t ro = (size_t)(mOff + mi * 16 + lr) * K + aC0;
    aHr[mi] = aHip + ro;
    aLr[mi] = aLop + ro;
  }
#pragma unroll
  for (int ni = 0; ni < 4; ++ni) {
    size_t wo = (size_t)(nOff + ni * 16 + lr) * K + bC0;
    wHr[ni] = wHip + wo;
    wLr[ni] = wLop + wo;
  }

  v8f acc[2][4];
  v8f zero = {0.f, 0.f, 0.f, 0.f, 0.f, 0.f, 0.f, 0.f};
#pragma unroll
  for (int mi = 0; mi < 2; ++mi)
#pragma unroll
    for (int ni = 0; ni < 4; ++ni) acc[mi][ni] = zero;

#pragma unroll
  for (int kb = 0; kb < K; kb += 32) {
    v16bf aH[2], aL[2];
#pragma unroll
    for (int mi = 0; mi < 2; ++mi) {
      v8bf h0 = *(const v8bf*)(aHr[mi] + kb);
      v8bf h1 = *(const v8bf*)(aHr[mi] + kb + 8);
      v8bf l0 = *(const v8bf*)(aLr[mi] + kb);
      v8bf l1 = *(const v8bf*)(aLr[mi] + kb + 8);
      aH[mi] = __builtin_shufflevector(h0, h1, 0,1,2,3,4,5,6,7,8,9,10,11,12,13,14,15);
      aL[mi] = __builtin_shufflevector(l0, l1, 0,1,2,3,4,5,6,7,8,9,10,11,12,13,14,15);
    }
#pragma unroll
    for (int ni = 0; ni < 4; ++ni) {
      v16bf bH = *(const v16bf*)(wHr[ni] + kb);
      v16bf bL = *(const v16bf*)(wLr[ni] + kb);
#pragma unroll
      for (int mi = 0; mi < 2; ++mi) {
        acc[mi][ni] = __builtin_amdgcn_wmma_f32_16x16x32_bf16(false, aH[mi], false, bH, (short)0, acc[mi][ni], false, false);
        acc[mi][ni] = __builtin_amdgcn_wmma_f32_16x16x32_bf16(false, aL[mi], false, bH, (short)0, acc[mi][ni], false, false);
        acc[mi][ni] = __builtin_amdgcn_wmma_f32_16x16x32_bf16(false, aH[mi], false, bL, (short)0, acc[mi][ni], false, false);
      }
    }
  }

#pragma unroll
  for (int ni = 0; ni < 4; ++ni) {
    const int col = nOff + ni * 16 + lr;
    const float bb = badj[col];
#pragma unroll
    for (int mi = 0; mi < 2; ++mi) {
#pragma unroll
      for (int r = 0; r < 8; ++r) {
        const int row = mOff + mi * 16 + half * 8 + r;   // ISA 32-bit C/D 16x16 layout
        z[(size_t)row * H_DIM + col] = acc[mi][ni][r] + bb;
      }
    }
  }
}

// ---------------- IndRNN scan: h_t = relu(z_t + u*h_{t-1}); emits bf16 hi/lo + BN partials ----------------
// 32-deep register double-buffer keeps 32 loads in flight along the serial h-chain;
// z reads are non-temporal (each read exactly once; keep L2 for activations).
#define SCAN_U 32
__global__ __launch_bounds__(256) void scan_kernel(
    const float* __restrict__ z, const float* __restrict__ uvec,
    bf16* __restrict__ aHi, bf16* __restrict__ aLo,
    float* __restrict__ partSum, float* __restrict__ partSq) {
  const int tid = blockIdx.x * 256 + threadIdx.x;   // tid = b*H + h
  const float uu = uvec[tid & (H_DIM - 1)];
  float hprev = 0.f, s = 0.f, q = 0.f;
  float cur[SCAN_U], nxt[SCAN_U];
#pragma unroll
  for (int j = 0; j < SCAN_U; ++j)
    cur[j] = __builtin_nontemporal_load(z + (size_t)j * NB + tid);
  for (int t0 = 0; t0 < T_DIM; t0 += SCAN_U) {
    if (t0 + SCAN_U < T_DIM) {
#pragma unroll
      for (int j = 0; j < SCAN_U; ++j)
        nxt[j] = __builtin_nontemporal_load(z + (size_t)(t0 + SCAN_U + j) * NB + tid);
    }
#pragma unroll
    for (int j = 0; j < SCAN_U; ++j) {
      float hv = fmaxf(cur[j] + uu * hprev, 0.f);
      hprev = hv; s += hv; q += hv * hv;
      bf16 hi = (bf16)hv;
      size_t idx = (size_t)(t0 + j) * NB + tid;
      aHi[idx] = hi;
      aLo[idx] = (bf16)(hv - (float)hi);
    }
#pragma unroll
    for (int j = 0; j < SCAN_U; ++j) cur[j] = nxt[j];
  }
  partSum[tid] = s;    // unique (b,h) per thread -> deterministic stats
  partSq[tid]  = q;
}

// ---------------- final: BN(layer5) folded into Wout; out[b,c] (tiny) ----------------
__global__ __launch_bounds__(512) void final_kernel(
    const bf16* __restrict__ aHi, const bf16* __restrict__ aLo,
    const float* __restrict__ partSum, const float* __restrict__ partSq,
    const float* __restrict__ gamma5, const float* __restrict__ beta5,
    const float* __restrict__ Wout, const float* __restrict__ bout,
    float* __restrict__ out) {
  __shared__ float sA[H_DIM];
  __shared__ float sC[H_DIM];
  const int t = threadIdx.x;
  {
    float s = 0.f, q = 0.f;
    for (int bb = 0; bb < B_DIM; ++bb) {
      s += partSum[bb * H_DIM + t];
      q += partSq[bb * H_DIM + t];
    }
    float mean = s * INV_N;
    float var  = q * INV_N - mean * mean;
    float a = gamma5[t] * rsqrtf(var + BN_EPS);
    sA[t] = a;
    sC[t] = beta5[t] - mean * a;
  }
  __syncthreads();
  if (t < B_DIM * C_DIM) {
    const int b = t / C_DIM, c = t % C_DIM;
    const size_t base = ((size_t)(T_DIM - 1) * B_DIM + b) * H_DIM;
    float acc = bout[c];
    for (int h = 0; h < H_DIM; ++h) {
      float sv = (float)aHi[base + h] + (float)aLo[base + h];
      acc += (sv * sA[h] + sC[h]) * Wout[c * H_DIM + h];
    }
    out[b * C_DIM + c] = acc;
  }
}

// ---------------- launch ----------------
extern "C" void kernel_launch(void* const* d_in, const int* in_sizes, int n_in,
                              void* d_out, int out_size, void* d_ws, size_t ws_size,
                              hipStream_t stream) {
  (void)in_sizes; (void)n_in; (void)out_size; (void)ws_size;
  const float* x     = (const float*)d_in[0];
  const float* W0    = (const float*)d_in[1];
  const float* b0    = (const float*)d_in[2];
  const float* Wh    = (const float*)d_in[3];
  const float* bh    = (const float*)d_in[4];
  const float* u     = (const float*)d_in[5];
  const float* gamma = (const float*)d_in[6];
  const float* beta  = (const float*)d_in[7];
  const float* Wout  = (const float*)d_in[8];
  const float* bout  = (const float*)d_in[9];

  char* ws = (char*)d_ws;
  const size_t OFF_Z    = 0;
  const size_t OFF_AHI  = OFF_Z    + (size_t)M_DIM * H_DIM * 4;   // z: 128 MB fp32
  const size_t OFF_ALO  = OFF_AHI  + (size_t)M_DIM * H_DIM * 2;   // act hi: 64 MB
  const size_t OFF_WHI  = OFF_ALO  + (size_t)M_DIM * H_DIM * 2;   // act lo: 64 MB
  const size_t OFF_WLO  = OFF_WHI  + (size_t)H_DIM * H_DIM * 2;
  const size_t OFF_BADJ = OFF_WLO  + (size_t)H_DIM * H_DIM * 2;
  const size_t OFF_PSUM = OFF_BADJ + (size_t)H_DIM * 4;
  const size_t OFF_PSQ  = OFF_PSUM + (size_t)NB * 4;

  float* zbuf    = (float*)(ws + OFF_Z);
  bf16*  aHi     = (bf16*)(ws + OFF_AHI);
  bf16*  aLo     = (bf16*)(ws + OFF_ALO);
  bf16*  wHi     = (bf16*)(ws + OFF_WHI);
  bf16*  wLo     = (bf16*)(ws + OFF_WLO);
  float* badj    = (float*)(ws + OFF_BADJ);
  float* partSum = (float*)(ws + OFF_PSUM);
  float* partSq  = (float*)(ws + OFF_PSQ);

  // x -> bf16 hi/lo (layer-0 GEMM input), [M, D] flat
  convx_kernel<<<(M_DIM * D_DIM) / 1024, 256, 0, stream>>>(x, aHi, aLo);

  for (int l = 0; l < L_DIM; ++l) {
    const int Din = (l == 0) ? D_DIM : H_DIM;
    const float* W  = (l == 0) ? W0 : (Wh + (size_t)(l - 1) * H_DIM * H_DIM);
    const float* bv = (l == 0) ? b0 : (bh + (size_t)(l - 1) * H_DIM);
    const float* gp = (l == 0) ? gamma : (gamma + (size_t)(l - 1) * H_DIM);
    const float* bp = (l == 0) ? beta  : (beta  + (size_t)(l - 1) * H_DIM);

    prep_kernel<<<H_DIM, 256, 0, stream>>>(W, bv, gp, bp, partSum, partSq,
                                           wHi, wLo, badj, Din, (l > 0) ? 1 : 0);
    if (Din == D_DIM) {
      gemm_kernel<D_DIM><<<dim3(M_DIM / 64, H_DIM / 256), 256, 0, stream>>>(
          aHi, aLo, wHi, wLo, badj, zbuf);
    } else {
      gemm_kernel<H_DIM><<<dim3(M_DIM / 64, H_DIM / 256), 256, 0, stream>>>(
          aHi, aLo, wHi, wLo, badj, zbuf);
    }
    scan_kernel<<<NB / 256, 256, 0, stream>>>(zbuf, u + (size_t)l * H_DIM,
                                              aHi, aLo, partSum, partSq);
  }

  final_kernel<<<1, 512, 0, stream>>>(aHi, aLo, partSum, partSq,
                                      gamma + (size_t)5 * H_DIM, beta + (size_t)5 * H_DIM,
                                      Wout, bout, (float*)d_out);
}
